// EmbLoss_v1_10453950398911
// MI455X (gfx1250) — matrix operand order repaired and checked by reading, not compiled, not run on hardware
//
#include <hip/hip_runtime.h>
#include <hip/hip_bf16.h>
#include <math.h>

typedef __attribute__((ext_vector_type(16))) _Float16 v16h;
typedef __attribute__((ext_vector_type(8)))  float    v8f;

#define HW        409600      // 640*640
#define NBATCH    8
#define NBIN      16
#define WPI       800         // waves per image
#define GPW       16          // 32-pixel groups per wave (800*512 = 409600)
#define WGPI      100         // workgroups per image (8 waves each)

// ws layout (floats)
#define P1_OFF    0                       // [8*800][16][6]  (sum0..3, cnt_k, cnt)
#define P2_OFF    614400                  // [8*800][16]     (agg partial)
#define MEAN_OFF  716800                  // [8][16][4]
#define CNT_OFF   717312                  // [8][16]

__device__ __forceinline__ _Float16 h_from_bits(unsigned short u) {
    union { _Float16 h; unsigned short u; } c; c.u = u; return c.h;
}

// ---------------- Pass 1: one-hot WMMA segment sums -------------------------
__global__ void __launch_bounds__(256)
k_pass1(const float* __restrict__ emb, const int* __restrict__ inst_in,
        const float* __restrict__ kern, const float* __restrict__ tmask,
        float* __restrict__ part1) {
    const int b    = blockIdx.y;
    const int lane = threadIdx.x & 31;
    const int wib  = threadIdx.x >> 5;
    const int w    = blockIdx.x * 8 + wib;            // 0..799
    const int slot = b * WPI + w;
    const int m    = lane & 15;                       // A row / B column / C column
    const bool hi  = lane >= 16;

    const float* e0p = emb + (size_t)b * 4 * HW;
    const float* e1p = e0p + HW;
    const float* e2p = e0p + 2 * HW;
    const float* e3p = e0p + 3 * HW;
    const int*   ip  = inst_in + (size_t)b * HW;
    const float* kp  = kern    + (size_t)b * HW;
    const float* tp  = tmask   + (size_t)b * HW;

    v8f c1 = {};   // onehot(inst_k) x [emb0..3, 1]
    v8f c2 = {};   // onehot(inst)   x [emb0..3, 1]

    for (int g = 0; g < GPW; ++g) {
        const int p = w * 512 + g * 32 + lane;        // pixel in image
        const float f0 = e0p[p], f1 = e1p[p], f2 = e2p[p], f3 = e3p[p];
        const int   iv = ip[p];
        const int   il = (tp[p] > 0.5f) ? iv : 0;     // inst
        const int   ik = (kp[p] > 0.5f) ? il : 0;     // inst_k
        const int   lbl = il | (ik << 8);

        union { _Float16 h[2]; int i; } u01, u23;
        u01.h[0] = (_Float16)f0; u01.h[1] = (_Float16)f1;
        u23.h[0] = (_Float16)f2; u23.h[1] = (_Float16)f3;
        const int h01 = u01.i, h23 = u23.i;

        v16h A1, A2, Bm;
        #pragma unroll
        for (int e = 0; e < 16; ++e) {
            // A-matrix gather: K = (e<8 ? e : e+8) + 8*hi
            const int aSrc = ((e < 8) ? e : e + 8) + (hi ? 8 : 0);
            const int s    = __shfl(lbl, aSrc, 32);
            A1[e] = (((s >> 8) & 0xFF) == m) ? (_Float16)1.0f : (_Float16)0.0f;
            A2[e] = ((s & 0xFF) == m)        ? (_Float16)1.0f : (_Float16)0.0f;
            // B-matrix gather: lane holds column m; element e <-> K = (e<8 ? e : e+8) + 8*hi
            const int bSrc = ((e < 8) ? e : e + 8) + (hi ? 8 : 0);
            const int s01  = __shfl(h01, bSrc, 32);
            const int s23  = __shfl(h23, bSrc, 32);
            const int sel  = (m < 2) ? s01 : s23;
            int hb = ((m & 1) ? (sel >> 16) : sel) & 0xFFFF;
            if (m >= 4) hb = (m == 4) ? 0x3C00 : 0;   // col4 = 1.0h (count), rest 0
            Bm[e] = h_from_bits((unsigned short)hb);
        }
        c1 = __builtin_amdgcn_wmma_f32_16x16x32_f16(false, A1, false, Bm, (short)0, c1, false, false);
        c2 = __builtin_amdgcn_wmma_f32_16x16x32_f16(false, A2, false, Bm, (short)0, c2, false, false);
    }

    // write per-wave partials: feat 0..3 = emb sums, 4 = cnt_k, 5 = cnt
    if (m <= 4) {
        #pragma unroll
        for (int r = 0; r < 8; ++r) {
            const int bin = r + (hi ? 8 : 0);
            part1[(size_t)slot * 96 + bin * 6 + m] = c1[r];
        }
    }
    if (m == 4) {
        #pragma unroll
        for (int r = 0; r < 8; ++r) {
            const int bin = r + (hi ? 8 : 0);
            part1[(size_t)slot * 96 + bin * 6 + 5] = c2[r];
        }
    }
}

// ---------------- Means: deterministic fixed-order reduction ----------------
__global__ void __launch_bounds__(128)
k_means(const float* __restrict__ part1, float* __restrict__ wsMean,
        float* __restrict__ wsCnt) {
    const int t = threadIdx.x;          // 0..127 : b = t>>4, bin = t&15
    const int b = t >> 4, bin = t & 15;
    float s0 = 0, s1 = 0, s2 = 0, s3 = 0, ck = 0, cn = 0;
    for (int w = 0; w < WPI; ++w) {
        const float* p = part1 + (size_t)(b * WPI + w) * 96 + bin * 6;
        s0 += p[0]; s1 += p[1]; s2 += p[2]; s3 += p[3]; ck += p[4]; cn += p[5];
    }
    const float d = fmaxf(ck, 1.0f);
    float m0 = s0 / d, m1 = s1 / d, m2 = s2 / d, m3 = s3 / d;
    if (bin == 0) { m0 = m1 = m2 = m3 = 0.0f; }
    float* mp = wsMean + (size_t)(b * NBIN + bin) * 4;
    mp[0] = m0; mp[1] = m1; mp[2] = m2; mp[3] = m3;
    wsCnt[b * NBIN + bin] = cn;
}

// ---------------- Pass 2: per-pixel loss, one-hot WMMA segment sum ----------
__global__ void __launch_bounds__(256)
k_pass2(const float* __restrict__ emb, const int* __restrict__ inst_in,
        const float* __restrict__ tmask, const float* __restrict__ wsMean,
        float* __restrict__ part2) {
    __shared__ __align__(16) float sMean[NBIN * 4];
    const int b = blockIdx.y;
    if (threadIdx.x < NBIN * 4) sMean[threadIdx.x] = wsMean[b * NBIN * 4 + threadIdx.x];
    __syncthreads();

    const int lane = threadIdx.x & 31;
    const int wib  = threadIdx.x >> 5;
    const int w    = blockIdx.x * 8 + wib;
    const int slot = b * WPI + w;
    const int m    = lane & 15;
    const bool hi  = lane >= 16;

    const float* e0p = emb + (size_t)b * 4 * HW;
    const float* e1p = e0p + HW;
    const float* e2p = e0p + 2 * HW;
    const float* e3p = e0p + 3 * HW;
    const int*   ip  = inst_in + (size_t)b * HW;
    const float* tp  = tmask   + (size_t)b * HW;

    v8f c = {};
    for (int g = 0; g < GPW; ++g) {
        const int p  = w * 512 + g * 32 + lane;
        const float f0 = e0p[p], f1 = e1p[p], f2 = e2p[p], f3 = e3p[p];
        const int   il = (tp[p] > 0.5f) ? ip[p] : 0;
        const float4 mu = *reinterpret_cast<const float4*>(&sMean[il * 4]);
        const float d0 = f0 - mu.x, d1 = f1 - mu.y, d2 = f2 - mu.z, d3 = f3 - mu.w;
        const float sq = d0 * d0 + d1 * d1 + d2 * d2 + d3 * d3;
        const float dist = (sq > 0.0f) ? sqrtf(sq) : 0.0f;
        float r = dist - 0.5f; r = (r > 0.0f) ? r : 0.0f;
        const float val = logf(r * r + 1.0f);

        union { _Float16 h[2]; int i; } uv;
        uv.h[0] = (_Float16)val; uv.h[1] = (_Float16)0.0f;
        const int hv = uv.i;

        v16h A, Bm;
        #pragma unroll
        for (int e = 0; e < 16; ++e) {
            const int aSrc = ((e < 8) ? e : e + 8) + (hi ? 8 : 0);
            const int s    = __shfl(il, aSrc, 32);
            A[e] = (s == m) ? (_Float16)1.0f : (_Float16)0.0f;
            const int bSrc = ((e < 8) ? e : e + 8) + (hi ? 8 : 0);
            const int sv   = __shfl(hv, bSrc, 32);
            const int hb   = (m == 0) ? (sv & 0xFFFF) : 0;
            Bm[e] = h_from_bits((unsigned short)hb);
        }
        c = __builtin_amdgcn_wmma_f32_16x16x32_f16(false, A, false, Bm, (short)0, c, false, false);
    }
    if (m == 0) {
        #pragma unroll
        for (int r = 0; r < 8; ++r) {
            const int bin = r + (hi ? 8 : 0);
            part2[(size_t)slot * NBIN + bin] = c[r];
        }
    }
}

// ---------------- Final scalar loss ----------------------------------------
__global__ void __launch_bounds__(128)
k_final(const float* __restrict__ part2, const float* __restrict__ wsMean,
        const float* __restrict__ wsCnt, float* __restrict__ out) {
    __shared__ float agg[NBATCH][NBIN];
    __shared__ float lossB[NBATCH];
    const int t = threadIdx.x;
    {
        const int b = t >> 4, k = t & 15;
        float s = 0.0f;
        for (int w = 0; w < WPI; ++w) s += part2[(size_t)(b * WPI + w) * NBIN + k];
        agg[b][k] = s;
    }
    __syncthreads();
    if (t < NBATCH) {
        const int b = t;
        float la = 0.0f;
        for (int k = 1; k < NBIN; ++k)
            la += agg[b][k] / fmaxf(wsCnt[b * NBIN + k], 1.0f);
        la /= 15.0f;

        float ld = 0.0f, lr = 0.0f;
        for (int i = 0; i < NBIN; ++i) {
            const float* mi = wsMean + (size_t)(b * NBIN + i) * 4;
            const float sqi = mi[0]*mi[0] + mi[1]*mi[1] + mi[2]*mi[2] + mi[3]*mi[3];
            const float ni  = (sqi > 0.0f) ? sqrtf(sqi) : 0.0f;
            lr += logf(ni + 1.0f);
            for (int j = 0; j < NBIN; ++j) {
                if (i == 0 || j == 0 || i == j) continue;
                const float* mj = wsMean + (size_t)(b * NBIN + j) * 4;
                const float d0 = mi[0]-mj[0], d1 = mi[1]-mj[1],
                            d2 = mi[2]-mj[2], d3 = mi[3]-mj[3];
                const float s2 = d0*d0 + d1*d1 + d2*d2 + d3*d3;
                const float pd = (s2 > 0.0f) ? sqrtf(s2) : 0.0f;
                float u = 3.0f - pd; u = (u > 0.0f) ? u : 0.0f;  // 2*DELTA_D - pd
                ld += logf(u * u + 1.0f);
            }
        }
        ld /= 210.0f;                  // (K-1)*(K-2)
        lr  = (lr / 16.0f) * 0.001f;
        lossB[b] = la + ld + lr;
    }
    __syncthreads();
    if (t == 0) {
        float s = 0.0f;
        for (int b = 0; b < NBATCH; ++b) s += lossB[b];
        out[0] = s / 8.0f;
    }
}

extern "C" void kernel_launch(void* const* d_in, const int* in_sizes, int n_in,
                              void* d_out, int out_size, void* d_ws, size_t ws_size,
                              hipStream_t stream) {
    (void)in_sizes; (void)n_in; (void)out_size; (void)ws_size;
    const float* emb   = (const float*)d_in[0];
    const int*   inst  = (const int*)  d_in[1];
    const float* kern  = (const float*)d_in[2];
    const float* tmsk  = (const float*)d_in[3];
    // d_in[4] (bboxes) unused, as in the reference.
    float* ws  = (float*)d_ws;
    float* out = (float*)d_out;

    float* part1  = ws + P1_OFF;
    float* part2  = ws + P2_OFF;
    float* wsMean = ws + MEAN_OFF;
    float* wsCnt  = ws + CNT_OFF;

    dim3 grid(WGPI, NBATCH);
    k_pass1<<<grid, 256, 0, stream>>>(emb, inst, kern, tmsk, part1);
    k_means<<<1, 128, 0, stream>>>(part1, wsMean, wsCnt);
    k_pass2<<<grid, 256, 0, stream>>>(emb, inst, tmsk, wsMean, part2);
    k_final<<<1, 128, 0, stream>>>(part2, wsMean, wsCnt, out);
}
